// ARLanguageModel_23854248362741
// MI455X (gfx1250) — matrix operand order repaired and checked by reading, not compiled, not run on hardware
//
#include <hip/hip_runtime.h>
#include <math.h>

// ---------------------------------------------------------------------------
// Model constants (match reference)
// ---------------------------------------------------------------------------
#define BB     4
#define LL     1024
#define DD     1024
#define HH     16
#define HDIM   64
#define NLAYER 8
#define VOCAB  32000

typedef __attribute__((ext_vector_type(16))) __bf16 v16bf;
typedef __attribute__((ext_vector_type(8)))  float  v8f;

union Frag {
    v16bf  v;
    uint4  u[2];
    __bf16 e[16];
};

__device__ __forceinline__ __bf16 f2bf(float f) {
    unsigned u = __float_as_uint(f);
    u += 0x7FFFu + ((u >> 16) & 1u);          // round-to-nearest-even
    unsigned short h = (unsigned short)(u >> 16);
    __bf16 r;
    __builtin_memcpy(&r, &h, 2);
    return r;
}

__device__ __forceinline__ v8f wmma_bf16(v16bf a, v16bf b, v8f c) {
#if defined(__HIP_DEVICE_COMPILE__)
    // (neg_a, A, neg_b, B, c_mod, C, reuse_a, reuse_b)
    return __builtin_amdgcn_wmma_f32_16x16x32_bf16(false, a, false, b,
                                                   (short)0, c, false, false);
#else
    (void)a; (void)b;
    return c;
#endif
}

// Async copy of 16 bytes global -> LDS (CDNA5 DMA path, ASYNCcnt-tracked).
__device__ __forceinline__ void async_g2l_b128(void* lds_dst, const void* gsrc) {
#if defined(__HIP_DEVICE_COMPILE__)
    unsigned loff = (unsigned)(uintptr_t)lds_dst;   // LDS aperture: addr[31:0]
    asm volatile("global_load_async_to_lds_b128 %0, %1, off"
                 :: "v"(loff), "v"(gsrc) : "memory");
#else
    (void)lds_dst; (void)gsrc;
#endif
}

__device__ __forceinline__ void wait_async_zero() {
#if defined(__HIP_DEVICE_COMPILE__)
    asm volatile("s_wait_asynccnt 0x0" ::: "memory");
#endif
}

// ---------------------------------------------------------------------------
// 1) Embedding gather + sinusoidal positional encoding  -> x fp32 [B,L,D]
// ---------------------------------------------------------------------------
__global__ __launch_bounds__(256)
void embed_pe_kernel(const int* __restrict__ idx,
                     const float* __restrict__ emb,
                     float* __restrict__ x) {
    size_t i = (size_t)blockIdx.x * 256 + threadIdx.x;   // over B*L*D
    int    d  = (int)(i % DD);
    size_t bl = i / DD;
    int    l  = (int)(bl % LL);
    int    tok = idx[bl];
    int    i2  = d & ~1;
    float  div = __expf((float)i2 * (-9.210340371976184f / (float)DD));
    float  ang = (float)l * div;
    float  pe  = (d & 1) ? __cosf(ang) : __sinf(ang);
    x[i] = emb[(size_t)tok * DD + d] + pe;
}

// ---------------------------------------------------------------------------
// 2) LayerNorm (one block per row of D=1024) -> bf16 output
// ---------------------------------------------------------------------------
__global__ __launch_bounds__(256)
void ln_bf16_kernel(const float* __restrict__ x,
                    const float* __restrict__ sc,
                    const float* __restrict__ bi,
                    __bf16* __restrict__ out) {
    int row  = blockIdx.x;
    int tid  = threadIdx.x;
    int lane = tid & 31;
    int wave = tid >> 5;
    const float* xr = x + (size_t)row * DD;
    __shared__ float rb[8];

    float acc = 0.f;
    for (int i = tid; i < DD; i += 256) acc += xr[i];
    for (int o = 16; o; o >>= 1) acc += __shfl_xor(acc, o, 32);
    if (lane == 0) rb[wave] = acc;
    __syncthreads();
    float mean = 0.f;
    for (int w = 0; w < 8; ++w) mean += rb[w];
    mean *= (1.0f / (float)DD);
    __syncthreads();

    float vac = 0.f;
    for (int i = tid; i < DD; i += 256) {
        float dlt = xr[i] - mean;
        vac += dlt * dlt;
    }
    for (int o = 16; o; o >>= 1) vac += __shfl_xor(vac, o, 32);
    if (lane == 0) rb[wave] = vac;
    __syncthreads();
    float var = 0.f;
    for (int w = 0; w < 8; ++w) var += rb[w];
    var *= (1.0f / (float)DD);
    float inv = rsqrtf(var + 1e-5f);

    __bf16* orow = out + (size_t)row * DD;
    for (int i = tid; i < DD; i += 256)
        orow[i] = f2bf((xr[i] - mean) * inv * sc[i] + bi[i]);
}

// ---------------------------------------------------------------------------
// 3) WMMA GEMM: out = epilogue( A[MxK](bf16) @ W[KxN](fp32->bf16) )
//    Workgroup tile 128x64, 8 waves each 32x32 (4 accumulators, fragment
//    reuse -> 4 WMMAs per K-step). Double-buffered LDS, async A-tile DMA.
// ---------------------------------------------------------------------------
__global__ __launch_bounds__(256)
void gemm_bf16_kernel(const __bf16* __restrict__ A,
                      const float* __restrict__ W,
                      const float* __restrict__ bias,    // nullable
                      const float* __restrict__ resid,   // nullable
                      float* __restrict__ outF,          // one of outF/outB
                      __bf16* __restrict__ outB,
                      int M, int N, int K, int applyGelu) {
    __shared__ __align__(16) __bf16 As[2][128][32];   // activations (M x K)
    __shared__ __align__(16) __bf16 WT[2][64][32];    // weights transposed (N x K)

    const int tid  = threadIdx.x;
    const int lane = tid & 31;
    const int wave = tid >> 5;
    const int wm   = wave & 3;      // 0..3 -> M sub-tile (32 rows)
    const int wn   = wave >> 2;     // 0..1 -> N sub-tile (32 cols)
    const int half = lane >> 4;
    const int lr   = lane & 15;

    const int m0 = blockIdx.y * 128;
    const int n0 = blockIdx.x * 64;

    v8f c00 = {}, c01 = {}, c10 = {}, c11 = {};

    // ---- staging helpers -------------------------------------------------
    auto stageA = [&](int p, int k0) {
        // 128x32 bf16 = 8 KB; 512 chunks of 16 B; 2 async DMAs per thread
#pragma unroll
        for (int t = 0; t < 2; ++t) {
            int ch = tid * 2 + t;
            int r  = ch >> 2;
            int cc = (ch & 3) * 8;
            async_g2l_b128(&As[p][r][cc],
                           A + (size_t)(m0 + r) * K + k0 + cc);
        }
    };
    auto stageW = [&](int p, int k0) {
        // 32(K) x 64(N) fp32 -> bf16, stored transposed [n][k]
        int kk = tid >> 3;
        int nn = (tid & 7) * 8;
        const float* src = W + (size_t)(k0 + kk) * N + n0 + nn;
        if (k0 + 64 < K)
            __builtin_prefetch(W + (size_t)(k0 + 64 + kk) * N + n0 + nn, 0, 1);
#pragma unroll
        for (int j = 0; j < 8; ++j) WT[p][nn + j][kk] = f2bf(src[j]);
    };
    auto loadA = [&](int p, int mrow) {
        Frag a;
        const __bf16* ar = &As[p][mrow + lr][0];
        a.u[0] = *(const uint4*)(ar + (half ? 8 : 0));
        a.u[1] = *(const uint4*)(ar + (half ? 24 : 16));
        return a;
    };
    auto loadB = [&](int p, int nbc) {
        Frag b;
        const __bf16* br = &WT[p][nbc + lr][half * 8];
        b.u[0] = *(const uint4*)br;
        b.u[1] = *(const uint4*)(br + 16);
        return b;
    };

    // ---- main loop (double-buffered, one barrier per K-step) -------------
    const int nk = K / 32;
    stageA(0, 0);
    stageW(0, 0);
    wait_async_zero();
    __syncthreads();

    for (int ks = 0; ks < nk; ++ks) {
        int p = ks & 1;
        if (ks + 1 < nk) {
            stageA(1 - p, (ks + 1) * 32);
            stageW(1 - p, (ks + 1) * 32);
        }
        Frag a0 = loadA(p, wm * 32);
        Frag a1 = loadA(p, wm * 32 + 16);
        Frag b0 = loadB(p, wn * 32);
        Frag b1 = loadB(p, wn * 32 + 16);
        c00 = wmma_bf16(a0.v, b0.v, c00);
        c01 = wmma_bf16(a0.v, b1.v, c01);
        c10 = wmma_bf16(a1.v, b0.v, c10);
        c11 = wmma_bf16(a1.v, b1.v, c11);
        wait_async_zero();
        __syncthreads();
    }

    // ---- epilogue. C layout: VGPR r -> row (half*8 + r), col lr ----------
    auto epi = [&](float vv, size_t gm, size_t gn) {
        if (bias) vv += bias[gn];
        if (applyGelu) vv = 0.5f * vv * (1.0f + erff(vv * 0.70710678118f));
        if (resid) vv += resid[gm * (size_t)N + gn];
        if (outF) outF[gm * (size_t)N + gn] = vv;
        else      outB[gm * (size_t)N + gn] = f2bf(vv);
    };
    size_t gmb = (size_t)m0 + wm * 32 + half * 8;
    size_t gnb = (size_t)n0 + wn * 32 + lr;
#pragma unroll
    for (int r = 0; r < 8; ++r) {
        epi(c00[r], gmb + r,      gnb);
        epi(c01[r], gmb + r,      gnb + 16);
        epi(c10[r], gmb + r + 16, gnb);
        epi(c11[r], gmb + r + 16, gnb + 16);
    }
}

// ---------------------------------------------------------------------------
// 4) Fused causal attention. One workgroup per (b, h, 16-row tile).
//    scores(16 x L) kept in LDS (64 KB; CDNA5 WGP has 320 KB).
// ---------------------------------------------------------------------------
__global__ __launch_bounds__(256)
void attn_kernel(const __bf16* __restrict__ q,
                 const __bf16* __restrict__ kmat,
                 const __bf16* __restrict__ vmat,
                 __bf16* __restrict__ y) {
    __shared__ float sc[16][LL];   // 64 KB

    const int bid = blockIdx.x;
    const int it  = bid % (LL / 16);
    const int h   = (bid / (LL / 16)) % HH;
    const int b   = bid / ((LL / 16) * HH);
    const int i0  = it * 16;

    const int tid  = threadIdx.x;
    const int lane = tid & 31;
    const int wave = tid >> 5;
    const int half = lane >> 4;
    const int lr   = lane & 15;

    // ---- scores: Q(16x64) @ K^T tiles, waves strided over j-tiles ----
    for (int jt = wave; jt <= it; jt += 8) {
        const int j0 = jt * 16;
        v8f c = {};
#pragma unroll
        for (int kk = 0; kk < HDIM; kk += 32) {
            Frag a, bf_;
            const __bf16* qrow =
                q + ((size_t)(b * LL + i0 + lr) * DD + h * HDIM + kk);
            a.u[0] = *(const uint4*)(qrow + (half ? 8 : 0));
            a.u[1] = *(const uint4*)(qrow + (half ? 24 : 16));
            // B(k=hd, n=j): lane half 0 -> hd kk+0..7,16..23; half 1 -> kk+8..15,24..31
            const __bf16* krow =
                kmat + ((size_t)(b * LL + j0 + lr) * DD + h * HDIM + kk + half * 8);
            bf_.u[0] = *(const uint4*)krow;
            bf_.u[1] = *(const uint4*)(krow + 16);
            c = wmma_bf16(a.v, bf_.v, c);
        }
#pragma unroll
        for (int r = 0; r < 8; ++r) {
            int   mi = half * 8 + r;       // local row
            int   nj = lr;                 // local col
            float sv = c[r] * 0.125f;      // 1/sqrt(64)
            if (jt == it && (j0 + nj) > (i0 + mi)) sv = -3.0e38f;
            sc[mi][j0 + nj] = sv;
        }
    }
    __syncthreads();

    // ---- softmax over rows, 16 threads per row ----
    {
        int row = tid >> 4;
        int sl  = tid & 15;
        int len = (it + 1) * 16;
        float mx = -3.0e38f;
        for (int j = sl; j < len; j += 16) mx = fmaxf(mx, sc[row][j]);
        for (int o = 8; o; o >>= 1) mx = fmaxf(mx, __shfl_xor(mx, o, 16));
        float sum = 0.f;
        for (int j = sl; j < len; j += 16) {
            float e = __expf(sc[row][j] - mx);
            sc[row][j] = e;
            sum += e;
        }
        for (int o = 8; o; o >>= 1) sum += __shfl_xor(sum, o, 16);
        float inv = 1.0f / sum;
        for (int j = sl; j < len; j += 16) sc[row][j] *= inv;
    }
    __syncthreads();

    // ---- y tile = P(16 x len) @ V(len x 64); 4 waves, 16-wide hd chunks ----
    if (wave < 4) {
        const int len = (it + 1) * 16;
        const int hd0 = wave * 16;
        v8f c = {};
        for (int j0 = 0; j0 < len; j0 += 32) {
            Frag a, bf_;
            int kb0 = j0 + (half ? 8 : 0);
            int kb1 = j0 + (half ? 24 : 16);
#pragma unroll
            for (int e = 0; e < 8; ++e) {
                int x0 = kb0 + e;
                int x1 = kb1 + e;
                a.e[e]     = (x0 < len) ? f2bf(sc[lr][x0]) : f2bf(0.f);
                a.e[8 + e] = (x1 < len) ? f2bf(sc[lr][x1]) : f2bf(0.f);
            }
#pragma unroll
            for (int e = 0; e < 16; ++e) {
                int jj = j0 + half * 8 + (e < 8 ? e : e + 8);
                bf_.e[e] = (jj < len)
                    ? vmat[(size_t)(b * LL + jj) * DD + h * HDIM + hd0 + lr]
                    : f2bf(0.f);
            }
            c = wmma_bf16(a.v, bf_.v, c);
        }
#pragma unroll
        for (int r = 0; r < 8; ++r) {
            int row = i0 + half * 8 + r;
            y[(size_t)(b * LL + row) * DD + h * HDIM + hd0 + lr] = f2bf(c[r]);
        }
    }
}

// ---------------------------------------------------------------------------
// Host-side orchestration
// ---------------------------------------------------------------------------
extern "C" void kernel_launch(void* const* d_in, const int* in_sizes, int n_in,
                              void* d_out, int out_size, void* d_ws, size_t ws_size,
                              hipStream_t stream) {
    (void)in_sizes; (void)n_in; (void)out_size; (void)ws_size;

    const int*   idx   = (const int*)  d_in[0];
    const float* emb   = (const float*)d_in[1];
    const float* ln1_s = (const float*)d_in[2];
    const float* ln1_b = (const float*)d_in[3];
    const float* wq    = (const float*)d_in[4];
    const float* wk    = (const float*)d_in[5];
    const float* wv    = (const float*)d_in[6];
    const float* wo    = (const float*)d_in[7];
    const float* ln2_s = (const float*)d_in[8];
    const float* ln2_b = (const float*)d_in[9];
    const float* w1    = (const float*)d_in[10];
    const float* b1    = (const float*)d_in[11];
    const float* w2    = (const float*)d_in[12];
    const float* b2    = (const float*)d_in[13];
    const float* lnf_s = (const float*)d_in[14];
    const float* lnf_b = (const float*)d_in[15];
    const float* headw = (const float*)d_in[16];

    const size_t M    = (size_t)BB * LL;          // 4096 rows
    const size_t MB_  = 1u << 20;
    char* ws = (char*)d_ws;
    float*  x  = (float*) (ws);                   // 16 MB fp32 residual
    __bf16* hb = (__bf16*)(ws + 16 * MB_);        //  8 MB
    __bf16* qb = (__bf16*)(ws + 24 * MB_);        //  8 MB
    __bf16* kb = (__bf16*)(ws + 32 * MB_);        //  8 MB
    __bf16* vb = (__bf16*)(ws + 40 * MB_);        //  8 MB
    __bf16* yb = (__bf16*)(ws + 48 * MB_);        //  8 MB
    __bf16* gb = (__bf16*)(ws + 56 * MB_);        // 32 MB (B*L*4D bf16)

    const size_t D2 = (size_t)DD * DD;
    const size_t DF = (size_t)DD * 4 * DD;

    embed_pe_kernel<<<(BB * LL * DD) / 256, 256, 0, stream>>>(idx, emb, x);

    dim3 gD (DD / 64,       (int)(M / 128));   // N = 1024
    dim3 gF (4 * DD / 64,   (int)(M / 128));   // N = 4096
    dim3 gV (VOCAB / 64,    (int)(M / 128));   // N = 32000

    for (int l = 0; l < NLAYER; ++l) {
        ln_bf16_kernel<<<(int)M, 256, 0, stream>>>(x, ln1_s + (size_t)l * DD,
                                                   ln1_b + (size_t)l * DD, hb);
        gemm_bf16_kernel<<<gD, 256, 0, stream>>>(hb, wq + (size_t)l * D2,
            nullptr, nullptr, nullptr, qb, (int)M, DD, DD, 0);
        gemm_bf16_kernel<<<gD, 256, 0, stream>>>(hb, wk + (size_t)l * D2,
            nullptr, nullptr, nullptr, kb, (int)M, DD, DD, 0);
        gemm_bf16_kernel<<<gD, 256, 0, stream>>>(hb, wv + (size_t)l * D2,
            nullptr, nullptr, nullptr, vb, (int)M, DD, DD, 0);

        attn_kernel<<<BB * HH * (LL / 16), 256, 0, stream>>>(qb, kb, vb, yb);

        gemm_bf16_kernel<<<gD, 256, 0, stream>>>(yb, wo + (size_t)l * D2,
            nullptr, x, x, nullptr, (int)M, DD, DD, 0);

        ln_bf16_kernel<<<(int)M, 256, 0, stream>>>(x, ln2_s + (size_t)l * DD,
                                                   ln2_b + (size_t)l * DD, hb);
        gemm_bf16_kernel<<<gF, 256, 0, stream>>>(hb, w1 + (size_t)l * DF,
            b1 + (size_t)l * 4 * DD, nullptr, nullptr, gb, (int)M, 4 * DD, DD, 1);
        gemm_bf16_kernel<<<gD, 256, 0, stream>>>(gb, w2 + (size_t)l * DF,
            b2 + (size_t)l * DD, x, x, nullptr, (int)M, DD, 4 * DD, 0);
    }

    ln_bf16_kernel<<<(int)M, 256, 0, stream>>>(x, lnf_s, lnf_b, hb);
    gemm_bf16_kernel<<<gV, 256, 0, stream>>>(hb, headw,
        nullptr, nullptr, (float*)d_out, nullptr, (int)M, VOCAB, DD, 0);
}
